// BRCLayer_8942121910696
// MI455X (gfx1250) — matrix operand order repaired and checked by reading, not compiled, not run on hardware
//
#include <hip/hip_runtime.h>

// ---------------------------------------------------------------------------
// BRC layer for MI455X (gfx1250, wave32, WMMA)
//   S=512, B=64, I=512, H=1024, M = S*B = 32768
// Plan:
//   1) cvt_x   : x fp32 -> bf16 (row-major [M][512])
//   2) cvt_w   : U[h][i] fp32 -> Wt[i][h] bf16 (K-major, matches WMMA B layout)
//   3) brc_gemm: pc/pa/ph[M][1024] f32 via v_wmma_f32_16x16x32_bf16
//      - all 12 B-fragments of a K-step staged in distinct regs so the
//        scheduler can clause the loads and overlap them with WMMAs
//      - loop-invariant base pointers + immediate IOFFSETs (no per-iter
//        64-bit address math)
//   4) brc_scan: per-(b,h) sequential recurrence over s, writes y and hn
// ---------------------------------------------------------------------------

#define SEQ_LEN 512
#define BATCH   64
#define ISZ     512
#define HSZ     1024
#define MROWS   (SEQ_LEN * BATCH)           // 32768
#define PROJ_ELEMS ((size_t)MROWS * HSZ)    // 33,554,432

typedef __attribute__((ext_vector_type(16))) __bf16 v16bf;
typedef __attribute__((ext_vector_type(8)))  __bf16 bf16x8;
typedef __attribute__((ext_vector_type(4)))  __bf16 bf16x4;
typedef __attribute__((ext_vector_type(8)))  float  v8f;
typedef __attribute__((ext_vector_type(4)))  float  f32x4;

union Frag { v16bf v; bf16x8 h[2]; };

// -------------------------- conversion kernels -----------------------------

__global__ void cvt_x_kernel(const float* __restrict__ x, __bf16* __restrict__ xb) {
    size_t i = ((size_t)blockIdx.x * blockDim.x + threadIdx.x) * 4;
    f32x4 v = *(const f32x4*)(x + i);
    *(bf16x4*)(xb + i) = __builtin_convertvector(v, bf16x4);
}

// U: [HSZ][ISZ] row-major fp32  ->  Wt: [ISZ][HSZ] bf16 (K-major)
__global__ void cvt_w_kernel(const float* __restrict__ U, __bf16* __restrict__ Wt) {
    int t = blockIdx.x * blockDim.x + threadIdx.x;   // 0 .. ISZ*HSZ-1
    int k = t >> 10;          // /HSZ
    int n = t & (HSZ - 1);
    Wt[t] = (__bf16)U[n * ISZ + k];
}

// ------------------------------- GEMM --------------------------------------
// Each wave: 16(M) x 64(N) tile of all three projections.
// grid = 4096 blocks x 256 threads (8 waves) -> 32768 wave-tiles
//   wt = blockIdx*8 + wave ; m-tile = wt>>4 ; n-tile = wt&15

__global__ void __launch_bounds__(256)
brc_gemm(const __bf16* __restrict__ xb,
         const __bf16* __restrict__ wt_c,
         const __bf16* __restrict__ wt_a,
         const __bf16* __restrict__ wt_h,
         float* __restrict__ pc,
         float* __restrict__ pa,
         float* __restrict__ ph)
{
    const int lane = threadIdx.x & 31;
    const int wave = threadIdx.x >> 5;
    const int wt   = blockIdx.x * 8 + wave;
    const int m0   = (wt >> 4) * 16;
    const int n0   = (wt & 15) * 64;

    v8f acc[3][4] = {};   // [matrix][n-subtile], 96 VGPRs of f32 accumulators

    // A-fragment addressing (16-bit A 16x32 layout, wave32):
    //   lanes 0-15 : row m0+lane,      K = {kb+0..7,  kb+16..23}
    //   lanes 16-31: row m0+(lane-16), K = {kb+8..15, kb+24..31}
    const __bf16* aptr =
        xb + (size_t)(m0 + (lane & 15)) * ISZ + ((lane >> 4) << 3);

    // B-fragment (16-bit B 32x16, wave32): lane <-> N = lane&15,
    // halves <-> K in same table order as A (strided rows of K-major Wt).
    // Base is loop-invariant; per-iteration displacement goes in IOFFSET
    // (max (503*1024 + 48)*2 B ~= 1030 KB < 2^23, fits signed 24-bit imm).
    const __bf16* const wp[3] = {
        wt_c + (size_t)((lane >> 4) << 3) * HSZ + n0 + (lane & 15),
        wt_a + (size_t)((lane >> 4) << 3) * HSZ + n0 + (lane & 15),
        wt_h + (size_t)((lane >> 4) << 3) * HSZ + n0 + (lane & 15)
    };

#pragma unroll
    for (int kb = 0; kb < ISZ; kb += 32) {
        Frag a;
        a.h[0] = *(const bf16x8*)(aptr + kb);
        a.h[1] = *(const bf16x8*)(aptr + kb + 16);

        // Stage all 12 B fragments first -> one deep load clause,
        // then 12 WMMAs drain with partial loadcnt waits.
        Frag b[3][4];
        const int ko = kb * HSZ;
#pragma unroll
        for (int m = 0; m < 3; ++m) {
#pragma unroll
            for (int j = 0; j < 4; ++j) {
#pragma unroll
                for (int e = 0; e < 8; ++e) {
                    b[m][j].h[0][e] = wp[m][ko + e * HSZ + j * 16];
                    b[m][j].h[1][e] = wp[m][ko + (16 + e) * HSZ + j * 16];
                }
            }
        }
#pragma unroll
        for (int m = 0; m < 3; ++m) {
#pragma unroll
            for (int j = 0; j < 4; ++j) {
                acc[m][j] = __builtin_amdgcn_wmma_f32_16x16x32_bf16(
                    false, a.v, false, b[m][j].v, (short)0, acc[m][j],
                    false, false);
            }
        }
    }

    // D layout: VGPR r, lanes 0-15 -> (m0+r, n0+lane); lanes 16-31 -> (m0+r+8,..)
    const size_t obase =
        (size_t)(m0 + ((lane >> 4) << 3)) * HSZ + n0 + (lane & 15);
    float* const outs[3] = { pc, pa, ph };
#pragma unroll
    for (int m = 0; m < 3; ++m) {
        float* o = outs[m] + obase;
#pragma unroll
        for (int j = 0; j < 4; ++j) {
#pragma unroll
            for (int r = 0; r < 8; ++r) {
                o[(size_t)r * HSZ + j * 16] = acc[m][j][r];
            }
        }
    }
}

// ------------------------------- scan --------------------------------------

__device__ __forceinline__ float fast_tanh(float x) {
#if __has_builtin(__builtin_amdgcn_tanhf)
    return __builtin_amdgcn_tanhf(x);
#elif __has_builtin(__builtin_amdgcn_tanh_f32)
    return __builtin_amdgcn_tanh_f32(x);
#else
    float e = __expf(2.0f * x);
    return 1.0f - 2.0f / (e + 1.0f);
#endif
}

__device__ __forceinline__ float fast_sigmoid(float x) {
    return 1.0f / (1.0f + __expf(-x));
}

__global__ void __launch_bounds__(256)
brc_scan(const float* __restrict__ h0,
         const float* __restrict__ w_c, const float* __restrict__ b_c,
         const float* __restrict__ w_a, const float* __restrict__ b_a,
         const float* __restrict__ b_h,
         const float* __restrict__ pc, const float* __restrict__ pa,
         const float* __restrict__ ph,
         float* __restrict__ y, float* __restrict__ hn)
{
    const int t   = blockIdx.x * blockDim.x + threadIdx.x;  // 0 .. B*H-1
    const int hid = t & (HSZ - 1);

    float h  = h0[t];
    const float wc = w_c[hid], bc = b_c[hid];
    const float wa = w_a[hid], ba = b_a[hid];
    const float bh = b_h[hid];

    size_t off = (size_t)t;
    for (int s = 0; s < SEQ_LEN; ++s, off += (size_t)BATCH * HSZ) {
        const float xc = pc[off] + bc;
        const float xa = pa[off] + ba;
        const float xh = ph[off] + bh;
        const float c  = fast_sigmoid(xc + wc * h);
        const float a  = 1.0f + fast_tanh(xa + wa * h);
        h = c * h + (1.0f - c) * fast_tanh(xh + a * h);
        y[off] = h;
    }
    hn[t] = h;
}

// ------------------------------ launcher -----------------------------------

extern "C" void kernel_launch(void* const* d_in, const int* in_sizes, int n_in,
                              void* d_out, int out_size, void* d_ws, size_t ws_size,
                              hipStream_t stream) {
    const float* x   = (const float*)d_in[0];
    const float* h0  = (const float*)d_in[1];
    const float* U_c = (const float*)d_in[2];
    const float* w_c = (const float*)d_in[3];
    const float* b_c = (const float*)d_in[4];
    const float* U_a = (const float*)d_in[5];
    const float* w_a = (const float*)d_in[6];
    const float* b_a = (const float*)d_in[7];
    const float* U_h = (const float*)d_in[8];
    const float* b_h = (const float*)d_in[9];

    // workspace layout (all offsets 256B-aligned by construction)
    char* w = (char*)d_ws;
    float*  pc   = (float*)w;                        // 128 MiB
    float*  pa   = pc + PROJ_ELEMS;                  // 128 MiB
    float*  ph   = pa + PROJ_ELEMS;                  // 128 MiB
    __bf16* xb   = (__bf16*)(ph + PROJ_ELEMS);       // 32 MiB
    __bf16* wt_c = xb + (size_t)MROWS * ISZ;         // 1 MiB each
    __bf16* wt_a = wt_c + (size_t)ISZ * HSZ;
    __bf16* wt_h = wt_a + (size_t)ISZ * HSZ;

    float* y  = (float*)d_out;
    float* hn = y + PROJ_ELEMS;

    // 1) x -> bf16  (16,777,216 elems, 4 per thread)
    cvt_x_kernel<<<(MROWS * ISZ) / (256 * 4), 256, 0, stream>>>(x, xb);

    // 2) U -> bf16 transposed (K-major), one launch per matrix
    cvt_w_kernel<<<(ISZ * HSZ) / 256, 256, 0, stream>>>(U_c, wt_c);
    cvt_w_kernel<<<(ISZ * HSZ) / 256, 256, 0, stream>>>(U_a, wt_a);
    cvt_w_kernel<<<(ISZ * HSZ) / 256, 256, 0, stream>>>(U_h, wt_h);

    // 3) fused triple GEMM: 32768 wave-tiles of 16x64, 8 waves per block
    brc_gemm<<<(MROWS / 16) * (HSZ / 64) / 8, 256, 0, stream>>>(
        xb, wt_c, wt_a, wt_h, pc, pa, ph);

    // 4) elementwise recurrence over s, one thread per (b,h)
    brc_scan<<<(BATCH * HSZ) / 256, 256, 0, stream>>>(
        h0, w_c, b_c, w_a, b_a, b_h, pc, pa, ph, y, hn);
}
